// Ranking_49959059587756
// MI455X (gfx1250) — matrix operand order repaired and checked by reading, not compiled, not run on hardware
//
#include <hip/hip_runtime.h>
#include <hip/hip_bf16.h>

// ---------------------------------------------------------------------------
// Problem constants (from the reference)
// ---------------------------------------------------------------------------
#define BB 8
#define NN 96
#define DD 256
#define HH 8
#define BIGROWS (BB * NN * NN)   // 73728, multiple of 32
#define QKVROWS (BB * NN)        // 768,   multiple of 32
#define DUROWS (BB * NN * 3)     // 2304,  multiple of 32

typedef __bf16 bf16_t;
typedef bf16_t v16bf __attribute__((ext_vector_type(16)));
typedef float  v8f   __attribute__((ext_vector_type(8)));

__device__ inline bf16_t f2bf(float f) {
  unsigned u = __builtin_bit_cast(unsigned, f);
  unsigned r = (u + 0x7FFFu + ((u >> 16) & 1u)) >> 16;
  unsigned short s = (unsigned short)r;
  return __builtin_bit_cast(bf16_t, s);
}

// fast silu: v_rcp path instead of IEEE divide
__device__ inline float silu_f(float x) {
  return __fdividef(x, 1.0f + __expf(-x));
}

// ---------------------------------------------------------------------------
// Fragment packing (CDNA5 wave32 WMMA 16x16x32 bf16)
//
// A (16x32, MxK): lane<16  holds M=lane,   i=0..7 -> K=i,     i=8..15 -> K=16+(i-8)
//                 lane>=16 holds M=lane-16,i=0..7 -> K=8+i,   i=8..15 -> K=24+(i-8)
// B (32x16, KxN): lane<16  holds N=lane,   i=0..7 -> K=i,     i=8..15 -> K=16+(i-8)
//                 lane>=16 holds N=lane-16,i=0..7 -> K=8+i,   i=8..15 -> K=24+(i-8)
// C (16x16): lane<16: rows 0..7 in c[0..7]; lane>=16: rows 8..15. col = lane&15.
// ---------------------------------------------------------------------------

// Stage a 32-row x 256-col fp32 tile (row stride 256) into LDS as packed
// bf16 A-fragments: two M-tiles of 8 k-fragments (512 bf16 each).
__device__ inline void stage_tile_f32_x2(const float* __restrict__ A,
                                         bf16_t* lds, long long R0) {
  for (int e = threadIdx.x; e < 8192; e += 512) {
    int row  = e >> 8;          // 0..31
    int kcol = e & 255;
    float v = A[(size_t)(R0 + row) * 256 + kcol];
    int tile = row >> 4, r = row & 15;
    int kt = kcol >> 5, kl = kcol & 31;
    int lane = r + (((kl >> 3) & 1) << 4);
    int i    = (kl & 7) + ((kl >> 4) << 3);
    lds[tile * 4096 + kt * 512 + lane * 16 + i] = f2bf(v);
  }
}

// 16 chained WMMAs: two 16-row accumulators share each B fragment.
__device__ inline void mma_k256_x2(const bf16_t* lds,
                                   const bf16_t* __restrict__ Wp, int ntGlobal,
                                   int l, v8f& c0, v8f& c1) {
#pragma unroll
  for (int kt = 0; kt < 8; ++kt) {
    v16bf b  = *(const v16bf*)(Wp + ((size_t)ntGlobal * 8 + kt) * 512 + l * 16);
    v16bf a0 = *(const v16bf*)(lds + kt * 512 + l * 16);
    v16bf a1 = *(const v16bf*)(lds + 4096 + kt * 512 + l * 16);
    c0 = __builtin_amdgcn_wmma_f32_16x16x32_bf16(false, a0, false, b, (short)0,
                                                 c0, false, false);
    c1 = __builtin_amdgcn_wmma_f32_16x16x32_bf16(false, a1, false, b, (short)0,
                                                 c1, false, false);
  }
}

// ---------------------------------------------------------------------------
// Pack an fp32 weight (256 x NC row-major) into bf16 WMMA B-fragments.
// ---------------------------------------------------------------------------
__global__ void convert_weight_kernel(const float* __restrict__ W,
                                      bf16_t* __restrict__ out, int NC) {
  int idx = blockIdx.x * 256 + threadIdx.x;
  if (idx >= 256 * NC) return;
  int fragIdx = idx >> 9;
  int within  = idx & 511;
  int lane = within >> 4, i = within & 15;
  int kt = fragIdx & 7, nt = fragIdx >> 3;
  int hi = lane >> 4;
  int kl = (i & 7) + (hi << 3) + ((i >> 3) << 4);  // B-matrix K layout
  int k  = kt * 32 + kl;
  int col = nt * 16 + (lane & 15);
  out[idx] = f2bf(W[(size_t)k * NC + col]);
}

// ---------------------------------------------------------------------------
// Generic fused GEMM: out(rows,NC) = [silu]( A(rows,256) @ W + bias )
// block: 512 threads = 16 waves; 32 rows x 256 cols per block.
// ---------------------------------------------------------------------------
__global__ __launch_bounds__(512) void gemm_kernel(
    const float* __restrict__ A, const bf16_t* __restrict__ Wp,
    const float* __restrict__ bias, float* __restrict__ out, int NC,
    int do_silu) {
  __shared__ __align__(32) bf16_t lds[8192];
  long long R0 = (long long)blockIdx.x * 32;
  stage_tile_f32_x2(A, lds, R0);
  __syncthreads();
  int wave = threadIdx.x >> 5, l = threadIdx.x & 31;
  int ntG = blockIdx.y * 16 + wave;
  int col = ntG * 16 + (l & 15);
  float b0 = bias ? bias[col] : 0.0f;
  v8f c0, c1;
#pragma unroll
  for (int j = 0; j < 8; ++j) { c0[j] = b0; c1[j] = b0; }
  mma_k256_x2(lds, Wp, ntG, l, c0, c1);
  int rhalf = (l >> 4) << 3;
#pragma unroll
  for (int j = 0; j < 8; ++j) {
    float v0 = c0[j], v1 = c1[j];
    if (do_silu) { v0 = silu_f(v0); v1 = silu_f(v1); }
    out[(size_t)(R0 + j + rhalf) * NC + col] = v0;
    out[(size_t)(R0 + 16 + j + rhalf) * NC + col] = v1;
  }
}

// ---------------------------------------------------------------------------
// Fused: dk = silu(edge_attr@Wdk+bdk) tile (never stored), immediately
// contracted with q,k into attn_weights[b,h,n,m]. Wave-xor reduce over the
// 16 cols a wave owns; atomicAdd into zero-initialized attn_w.
// 32 rows per block share (b,n) since 96 % 32 == 0.
// ---------------------------------------------------------------------------
__global__ __launch_bounds__(512) void dk_attnw_kernel(
    const float* __restrict__ edge_attr, const bf16_t* __restrict__ Wdk,
    const float* __restrict__ bdk, const float* __restrict__ q,
    const float* __restrict__ k, float* __restrict__ attn_w) {
  __shared__ __align__(32) bf16_t lds[8192];
  long long R0 = (long long)blockIdx.x * 32;
  stage_tile_f32_x2(edge_attr, lds, R0);
  __syncthreads();
  int wave = threadIdx.x >> 5, l = threadIdx.x & 31;
  int col = wave * 16 + (l & 15);
  float b0 = bdk[col];
  v8f c0, c1;
#pragma unroll
  for (int j = 0; j < 8; ++j) { c0[j] = b0; c1[j] = b0; }
  mma_k256_x2(lds, Wdk, wave, l, c0, c1);

  int b   = (int)(R0 / (NN * NN));
  int rem = (int)(R0 - (long long)b * NN * NN);
  int n = rem / NN, m0 = rem % NN;
  int h = wave >> 1;
  float qv = q[((size_t)(b * NN + n) << 8) + col];
  int rhalf = (l >> 4) << 3;
  float* awrow = attn_w + ((size_t)((b * HH + h) * NN + n) * NN);
#pragma unroll
  for (int tile = 0; tile < 2; ++tile) {
#pragma unroll
    for (int j = 0; j < 8; ++j) {
      int m = m0 + tile * 16 + j + rhalf;
      float acc = tile ? c1[j] : c0[j];
      float t = silu_f(acc) * qv * k[((size_t)(b * NN + m) << 8) + col];
      t += __shfl_xor(t, 1, 32);
      t += __shfl_xor(t, 2, 32);
      t += __shfl_xor(t, 4, 32);
      t += __shfl_xor(t, 8, 32);
      if ((l & 15) == 0) atomicAdd(&awrow[m], t);
    }
  }
}

// ---------------------------------------------------------------------------
// probs = silu(mask? 0 : attn_w) * cosine-cutoff(dist)
// ---------------------------------------------------------------------------
__global__ void probs_kernel(const float* __restrict__ aw,
                             const unsigned char* __restrict__ mask,
                             const float* __restrict__ dist,
                             float* __restrict__ probs) {
  int idx = blockIdx.x * 256 + threadIdx.x;  // ((b*H+h)*N+n)*N+m
  if (idx >= BB * HH * NN * NN) return;
  int m = idx % NN;
  int t = idx / NN;
  int n = t % NN;
  t /= NN;
  int b = t / HH;
  int bnm = (b * NN + n) * NN + m;
  float x = mask[bnm] ? 0.0f : aw[idx];
  float ds = dist[bnm];
  float sc = (ds < 5.0f) ? 0.5f * (__cosf(ds * 0.6283185307179586f) + 1.0f)
                         : 0.0f;
  probs[idx] = silu_f(x) * sc;
}

// ---------------------------------------------------------------------------
// attn[b,n,j] = sum_m probs[b, j>>5, n, m] * v[b,m,j]
// ---------------------------------------------------------------------------
__global__ void attn_sum_kernel(const float* __restrict__ probs,
                                const float* __restrict__ v,
                                float* __restrict__ attn) {
  int idx = blockIdx.x * 256 + threadIdx.x;  // (b*N+n)*256+col
  if (idx >= QKVROWS * DD) return;
  int col = idx & 255;
  int bn = idx >> 8;
  int n = bn % NN, b = bn / NN;
  int h = col >> 5;
  const float* pr = probs + (size_t)((b * HH + h) * NN + n) * NN;
  const float* vb = v + ((size_t)(b * NN) << 8) + col;
  float s = 0.0f;
  for (int m = 0; m < NN; ++m) s += pr[m] * vb[(size_t)m << 8];
  attn[idx] = s;
}

// ---------------------------------------------------------------------------
// Fused: A tile = attn_per_nodes (probs*v, composed on the fly, never stored),
// du_in = A@Wdu+bdu (tile, never stored), masked, contracted with vec into
// du[b,n,c,d] via cross-half shuffle + atomicAdd (du zero-initialized).
// ---------------------------------------------------------------------------
__global__ __launch_bounds__(512) void duin_du_kernel(
    const float* __restrict__ probs, const float* __restrict__ v,
    const bf16_t* __restrict__ Wdu, const float* __restrict__ bdu,
    const unsigned char* __restrict__ mask, const float* __restrict__ vec,
    float* __restrict__ du) {
  __shared__ __align__(32) bf16_t lds[8192];
  long long R0 = (long long)blockIdx.x * 32;
  int b   = (int)(R0 / (NN * NN));
  int rem = (int)(R0 - (long long)b * NN * NN);
  int n = rem / NN, m0 = rem % NN;

  for (int e = threadIdx.x; e < 8192; e += 512) {
    int row = e >> 8, kcol = e & 255;
    int m = m0 + row;
    float val = probs[((size_t)((b * HH + (kcol >> 5)) * NN + n) * NN) + m] *
                v[((size_t)(b * NN + m) << 8) + kcol];
    int tile = row >> 4, r = row & 15;
    int kt = kcol >> 5, kl = kcol & 31;
    int lane = r + (((kl >> 3) & 1) << 4);
    int i = (kl & 7) + ((kl >> 4) << 3);
    lds[tile * 4096 + kt * 512 + lane * 16 + i] = f2bf(val);
  }
  __syncthreads();

  int wave = threadIdx.x >> 5, l = threadIdx.x & 31;
  int col = wave * 16 + (l & 15);
  float b0 = bdu[col];
  v8f c0, c1;
#pragma unroll
  for (int j = 0; j < 8; ++j) { c0[j] = b0; c1[j] = b0; }
  mma_k256_x2(lds, Wdu, wave, l, c0, c1);

  int rhalf = (l >> 4) << 3;
  const unsigned char* mrow = mask + (size_t)(b * NN + n) * NN;
  const float* vrow = vec + (size_t)(b * NN + n) * NN * 3;
  float vals[16];
#pragma unroll
  for (int tile = 0; tile < 2; ++tile) {
#pragma unroll
    for (int j = 0; j < 8; ++j) {
      int m = m0 + tile * 16 + j + rhalf;
      float acc = tile ? c1[j] : c0[j];
      vals[tile * 8 + j] = mrow[m] ? 0.0f : acc;
    }
  }
#pragma unroll
  for (int ci = 0; ci < 3; ++ci) {
    float p = 0.0f;
#pragma unroll
    for (int tile = 0; tile < 2; ++tile) {
#pragma unroll
      for (int j = 0; j < 8; ++j) {
        int m = m0 + tile * 16 + j + rhalf;
        p += vals[tile * 8 + j] * vrow[(size_t)m * 3 + ci];
      }
    }
    p += __shfl_xor(p, 16, 32);  // combine rows 0-7 with rows 8-15
    if (l < 16)
      atomicAdd(&du[((size_t)((b * NN + n) * 3 + ci) << 8) + col], p);
  }
}

// ---------------------------------------------------------------------------
// Fused: ea = silu(edge_attr@Wea+bea) tile, multiplied elementwise by
// g[b,n,m,d] = sum_c w[b,n,c,d] * w[b,m,c,256+d]; written straight to ipe.
// ---------------------------------------------------------------------------
__global__ __launch_bounds__(512) void ipe_kernel(
    const float* __restrict__ edge_attr, const bf16_t* __restrict__ Wea,
    const float* __restrict__ bea, const float* __restrict__ w,
    float* __restrict__ ipe) {
  __shared__ __align__(32) bf16_t lds[8192];
  long long R0 = (long long)blockIdx.x * 32;
  stage_tile_f32_x2(edge_attr, lds, R0);
  __syncthreads();
  int wave = threadIdx.x >> 5, l = threadIdx.x & 31;
  int col = wave * 16 + (l & 15);
  float b0 = bea[col];
  v8f c0, c1;
#pragma unroll
  for (int j = 0; j < 8; ++j) { c0[j] = b0; c1[j] = b0; }
  mma_k256_x2(lds, Wea, wave, l, c0, c1);

  int b   = (int)(R0 / (NN * NN));
  int rem = (int)(R0 - (long long)b * NN * NN);
  int n = rem / NN, m0 = rem % NN;
  float wsv[3];
#pragma unroll
  for (int ci = 0; ci < 3; ++ci)
    wsv[ci] = w[((size_t)((b * NN + n) * 3 + ci) << 9) + col];
  int rhalf = (l >> 4) << 3;
#pragma unroll
  for (int tile = 0; tile < 2; ++tile) {
#pragma unroll
    for (int j = 0; j < 8; ++j) {
      int m = m0 + tile * 16 + j + rhalf;
      float g = 0.0f;
#pragma unroll
      for (int ci = 0; ci < 3; ++ci)
        g += wsv[ci] * w[((size_t)((b * NN + m) * 3 + ci) << 9) + 256 + col];
      float acc = tile ? c1[j] : c0[j];
      ipe[(size_t)(R0 + tile * 16 + j + rhalf) * DD + col] = silu_f(acc) * g;
    }
  }
}

// ---------------------------------------------------------------------------
// Host-side orchestration
// ---------------------------------------------------------------------------
extern "C" void kernel_launch(void* const* d_in, const int* in_sizes, int n_in,
                              void* d_out, int out_size, void* d_ws,
                              size_t ws_size, hipStream_t stream) {
  (void)in_sizes; (void)n_in; (void)out_size; (void)ws_size;
  const float* x         = (const float*)d_in[0];
  const float* vec       = (const float*)d_in[1];
  const float* dist      = (const float*)d_in[2];
  const float* edge_attr = (const float*)d_in[3];
  const unsigned char* mask = (const unsigned char*)d_in[4];
  const float* Wq  = (const float*)d_in[5];
  const float* bq  = (const float*)d_in[6];
  const float* Wk  = (const float*)d_in[7];
  const float* bk  = (const float*)d_in[8];
  const float* Wv  = (const float*)d_in[9];
  const float* bv  = (const float*)d_in[10];
  const float* Wdk = (const float*)d_in[11];
  const float* bdk = (const float*)d_in[12];
  const float* Wdu = (const float*)d_in[13];
  const float* bdu = (const float*)d_in[14];
  const float* Wdih = (const float*)d_in[15];
  const float* Wea  = (const float*)d_in[16];
  const float* bea  = (const float*)d_in[17];

  char* ws = (char*)d_ws;
  // byte offsets in workspace (all 256-aligned)
  const size_t OFF_WQ   = 0;            // 256*256 bf16 = 131072 B each
  const size_t OFF_WK   = 131072;
  const size_t OFF_WV   = 262144;
  const size_t OFF_WDK  = 393216;
  const size_t OFF_WDU  = 524288;
  const size_t OFF_WEA  = 655360;
  const size_t OFF_WDIH = 786432;       // 256*512 bf16 = 262144 B
  const size_t OFF_Q    = 1048576;      // 768*256 f32 = 786432 B
  const size_t OFF_K    = 1835008;
  const size_t OFF_V    = 2621440;
  const size_t OFF_AW   = 3407872;      // 589824 f32 = 2359296 B
  const size_t OFF_PR   = 5767168;
  const size_t OFF_DU   = 8126464;      // 589824 f32
  const size_t OFF_W    = 10485760;     // 1179648 f32 = 4718592 B

  bf16_t* wq_bf   = (bf16_t*)(ws + OFF_WQ);
  bf16_t* wk_bf   = (bf16_t*)(ws + OFF_WK);
  bf16_t* wv_bf   = (bf16_t*)(ws + OFF_WV);
  bf16_t* wdk_bf  = (bf16_t*)(ws + OFF_WDK);
  bf16_t* wdu_bf  = (bf16_t*)(ws + OFF_WDU);
  bf16_t* wea_bf  = (bf16_t*)(ws + OFF_WEA);
  bf16_t* wdih_bf = (bf16_t*)(ws + OFF_WDIH);
  float* q      = (float*)(ws + OFF_Q);
  float* k      = (float*)(ws + OFF_K);
  float* v      = (float*)(ws + OFF_V);
  float* attn_w = (float*)(ws + OFF_AW);
  float* probs  = (float*)(ws + OFF_PR);
  float* du     = (float*)(ws + OFF_DU);
  float* wbuf   = (float*)(ws + OFF_W);

  float* attn_out = (float*)d_out;                         // (B,N,D)
  float* ipe_out  = (float*)d_out + (size_t)QKVROWS * DD;  // (B,N,N,D)

  // 1) pack weights to bf16 WMMA fragments
  convert_weight_kernel<<<256, 256, 0, stream>>>(Wq, wq_bf, 256);
  convert_weight_kernel<<<256, 256, 0, stream>>>(Wk, wk_bf, 256);
  convert_weight_kernel<<<256, 256, 0, stream>>>(Wv, wv_bf, 256);
  convert_weight_kernel<<<256, 256, 0, stream>>>(Wdk, wdk_bf, 256);
  convert_weight_kernel<<<256, 256, 0, stream>>>(Wdu, wdu_bf, 256);
  convert_weight_kernel<<<256, 256, 0, stream>>>(Wea, wea_bf, 256);
  convert_weight_kernel<<<512, 256, 0, stream>>>(Wdih, wdih_bf, 512);

  // 2) zero accumulators
  hipMemsetAsync(attn_w, 0, 2359296, stream);
  hipMemsetAsync(du, 0, 2359296, stream);

  // 3) q, k, v projections
  gemm_kernel<<<dim3(QKVROWS / 32, 1), 512, 0, stream>>>(x, wq_bf, bq, q, 256, 0);
  gemm_kernel<<<dim3(QKVROWS / 32, 1), 512, 0, stream>>>(x, wk_bf, bk, k, 256, 0);
  gemm_kernel<<<dim3(QKVROWS / 32, 1), 512, 0, stream>>>(x, wv_bf, bv, v, 256, 0);

  // 4) dk GEMM fused with attention-weight contraction
  dk_attnw_kernel<<<BIGROWS / 32, 512, 0, stream>>>(edge_attr, wdk_bf, bdk, q, k,
                                                    attn_w);

  // 5) attention probabilities (mask, silu, cosine cutoff)
  probs_kernel<<<(BB * HH * NN * NN) / 256, 256, 0, stream>>>(attn_w, mask, dist,
                                                              probs);

  // 6) attn = sum_m probs * v  -> first output
  attn_sum_kernel<<<(QKVROWS * DD) / 256, 256, 0, stream>>>(probs, v, attn_out);

  // 7) du_in GEMM (A composed on the fly) fused with vec contraction -> du
  duin_du_kernel<<<BIGROWS / 32, 512, 0, stream>>>(probs, v, wdu_bf, bdu, mask,
                                                   vec, du);

  // 8) w = du @ Wdih  (rows 2304, cols 512)
  gemm_kernel<<<dim3(DUROWS / 32, 2), 512, 0, stream>>>(du, wdih_bf, nullptr,
                                                        wbuf, 512, 0);

  // 9) ipe = silu(edge_attr@Wea+bea) * einsum(ws,wt) -> second output
  ipe_kernel<<<BIGROWS / 32, 512, 0, stream>>>(edge_attr, wea_bf, bea, wbuf,
                                               ipe_out);
}
